// Codebook_4612794876163
// MI455X (gfx1250) — matrix operand order repaired and checked by reading, not compiled, not run on hardware
//
#include <hip/hip_runtime.h>
#include <hip/hip_bf16.h>
#include <math.h>

#define DDIM 512
#define WAVES_PER_BLOCK 8
#define ROWS_PER_WAVE 16
#define ROWS_PER_BLOCK (WAVES_PER_BLOCK * ROWS_PER_WAVE)  // 128
#define BTILE_HALVES (16 * DDIM)                           // one 16-code tile, bf16
#define BTILE_BYTES (BTILE_HALVES * 2)                     // 16 KB
#define KCHUNKS (DDIM / 32)                                // 16

typedef __attribute__((ext_vector_type(16))) __bf16 v16bf;
typedef __attribute__((ext_vector_type(8)))  float  v8f;

__device__ __forceinline__ unsigned short f32_to_bf16_rne(float f) {
    union { float f; unsigned int u; } v; v.f = f;
    unsigned int u = v.u;
    u += 0x7FFFu + ((u >> 16) & 1u);
    return (unsigned short)(u >> 16);
}

__device__ __forceinline__ float wave_reduce_sum(float x) {
    #pragma unroll
    for (int off = 16; off >= 1; off >>= 1) x += __shfl_xor(x, off, 32);
    return x;
}

// async-copy one 16-code B tile (16 KB) to LDS, spread over the 256-thread block:
// each thread issues 4 GLOBAL_LOAD_ASYNC_TO_LDS_B128 (ASYNCcnt += 4 per wave)
__device__ __forceinline__ void async_copy_b_tile(const unsigned short* __restrict__ embb,
                                                  int nt, unsigned lds_byte_base, int tid) {
    const unsigned long long gbase =
        (unsigned long long)(size_t)embb + (unsigned long long)nt * (unsigned long long)BTILE_BYTES;
    #pragma unroll
    for (int j = 0; j < 4; ++j) {
        const unsigned dst = lds_byte_base + (unsigned)(tid + 256 * j) * 16u;
        const unsigned long long src = gbase + (unsigned long long)(tid + 256 * j) * 16u;
        asm volatile("global_load_async_to_lds_b128 %0, %1, off"
                     :: "v"(dst), "v"(src)
                     : "memory");
    }
}

// ---------------- prep: normalize z rows, emit bf16 A operand ----------------
__global__ __launch_bounds__(256) void prep_z_kernel(const float* __restrict__ z,
                                                     unsigned short* __restrict__ znb) {
    const int wave = threadIdx.x >> 5;
    const int lane = threadIdx.x & 31;
    const int row  = blockIdx.x * WAVES_PER_BLOCK + wave;
    const float4* zr4 = (const float4*)(z + (size_t)row * DDIM + lane * 16);
    float4 v[4];
    float ss = 0.f;
    #pragma unroll
    for (int i = 0; i < 4; ++i) {
        v[i] = zr4[i];
        ss += v[i].x * v[i].x + v[i].y * v[i].y + v[i].z * v[i].z + v[i].w * v[i].w;
    }
    ss = wave_reduce_sum(ss);
    const float inv = 1.f / fmaxf(sqrtf(ss), 1e-12f);
    unsigned short h[16];
    #pragma unroll
    for (int i = 0; i < 4; ++i) {
        h[4*i+0] = f32_to_bf16_rne(v[i].x * inv);
        h[4*i+1] = f32_to_bf16_rne(v[i].y * inv);
        h[4*i+2] = f32_to_bf16_rne(v[i].z * inv);
        h[4*i+3] = f32_to_bf16_rne(v[i].w * inv);
    }
    int4* dst = (int4*)(znb + (size_t)row * DDIM + lane * 16);
    dst[0] = *(const int4*)&h[0];
    dst[1] = *(const int4*)&h[8];
}

// ---------------- prep: codebook rows -> bf16 B operand + 1/||e|| ----------------
__global__ __launch_bounds__(256) void prep_e_kernel(const float* __restrict__ emb,
                                                     unsigned short* __restrict__ embb,
                                                     float* __restrict__ inv_enorm) {
    const int wave = threadIdx.x >> 5;
    const int lane = threadIdx.x & 31;
    const int row  = blockIdx.x * WAVES_PER_BLOCK + wave;
    const float4* er4 = (const float4*)(emb + (size_t)row * DDIM + lane * 16);
    float4 v[4];
    float ss = 0.f;
    #pragma unroll
    for (int i = 0; i < 4; ++i) {
        v[i] = er4[i];
        ss += v[i].x * v[i].x + v[i].y * v[i].y + v[i].z * v[i].z + v[i].w * v[i].w;
    }
    ss = wave_reduce_sum(ss);
    unsigned short h[16];
    #pragma unroll
    for (int i = 0; i < 4; ++i) {
        h[4*i+0] = f32_to_bf16_rne(v[i].x);
        h[4*i+1] = f32_to_bf16_rne(v[i].y);
        h[4*i+2] = f32_to_bf16_rne(v[i].z);
        h[4*i+3] = f32_to_bf16_rne(v[i].w);
    }
    int4* dst = (int4*)(embb + (size_t)row * DDIM + lane * 16);
    dst[0] = *(const int4*)&h[0];
    dst[1] = *(const int4*)&h[8];
    if (lane == 0) inv_enorm[row] = 1.f / fmaxf(sqrtf(ss), 1e-8f);
}

// ---------------- bf16 WMMA similarity GEMM + fused argmax over codes ----------------
// block = 8 waves x 16 rows; B tiles shared block-wide via double-buffered async LDS copies;
// per-lane A slice held in 128 VGPRs across the whole code loop
__global__ __launch_bounds__(256) void argmax_gemm_kernel(const unsigned short* __restrict__ znb,
                                                          const unsigned short* __restrict__ embb,
                                                          const float* __restrict__ inv_enorm,
                                                          float* __restrict__ out_idx_f,
                                                          int* __restrict__ idx_i,
                                                          int Kc) {
    extern __shared__ unsigned short lds[];
    // layout (halves): [0, 128*512)            A tile  (128 KB)
    //                  [128*512, +2*16*512)    B double buffer (2 x 16 KB)
    const int tid  = threadIdx.x;
    const int wave = tid >> 5;
    const int lane = tid & 31;
    const int g    = lane >> 4;   // half-wave group (0/1)
    const int ln   = lane & 15;
    const int m_base   = blockIdx.x * ROWS_PER_BLOCK + wave * ROWS_PER_WAVE;
    const int numTiles = Kc / 16;

    // stage this wave's 16 A-rows into LDS (bf16)
    {
        const int4* src = (const int4*)(znb + (size_t)m_base * DDIM);
        int4* dst = (int4*)(lds + wave * ROWS_PER_WAVE * DDIM);
        #pragma unroll
        for (int i = lane; i < ROWS_PER_WAVE * DDIM / 8; i += 32) dst[i] = src[i];
    }

    // LDS byte offsets of the two B buffers (flat LDS aperture: low 32 bits = LDS offset)
    const unsigned lds_base  = (unsigned)(size_t)(void*)lds;
    const unsigned bbuf_byte[2] = { lds_base + (unsigned)(ROWS_PER_BLOCK * DDIM * 2),
                                    lds_base + (unsigned)(ROWS_PER_BLOCK * DDIM * 2) + BTILE_BYTES };

    // prologue: fill both B buffers
    async_copy_b_tile(embb, 0, bbuf_byte[0], tid);
    if (numTiles > 1) async_copy_b_tile(embb, 1, bbuf_byte[1], tid);

    // preload this lane's A slice into registers (loop-invariant across all code tiles)
    // A layout: lane holds row M = lane%16; halves 0-7 = K kb+8g.., halves 8-15 = K kb+16+8g..
    union AV { struct { int4 lo, hi; } p; v16bf v; };
    const unsigned short* Abase = lds + wave * ROWS_PER_WAVE * DDIM + ln * DDIM;
    AV areg[KCHUNKS];
    #pragma unroll
    for (int kc = 0; kc < KCHUNKS; ++kc) {
        areg[kc].p.lo = *(const int4*)(Abase + kc * 32 + 8 * g);
        areg[kc].p.hi = *(const int4*)(Abase + kc * 32 + 16 + 8 * g);
    }

    float best[8];
    int   bidx[8];
    #pragma unroll
    for (int i = 0; i < 8; ++i) { best[i] = -3.4e38f; bidx[i] = 0; }

    for (int nt = 0; nt < numTiles; ++nt) {
        const int buf = nt & 1;
        // retire the async batch for tile nt (batches complete in order, 4 per tile)
        if (nt + 1 < numTiles) asm volatile("s_wait_asynccnt 0x4" ::: "memory");
        else                   asm volatile("s_wait_asynccnt 0x0" ::: "memory");
        __syncthreads();   // all waves' async slices of this buffer have landed

        const int   n_base = nt * 16;
        const int   code   = n_base + ln;        // C layout: N = lane%16
        const float inv    = inv_enorm[code];    // issue early; overlaps the WMMA chain

        // B layout: lane holds column N = lane%16; halves 0-7 = K kb+8g.., halves 8-15 = K kb+16+8g..
        const unsigned short* Bbuf =
            lds + ROWS_PER_BLOCK * DDIM + buf * BTILE_HALVES + ln * DDIM + 8 * g;

        // two-deep ping-pong on B so ds_loads for chunk kc+1 overlap the WMMA of chunk kc
        union BV { struct { int4 lo, hi; } p; v16bf v; };
        BV b[2];
        b[0].p.lo = *(const int4*)(Bbuf + 0);
        b[0].p.hi = *(const int4*)(Bbuf + 16);

        v8f acc = {};
        #pragma unroll
        for (int kc = 0; kc < KCHUNKS; ++kc) {
            if (kc + 1 < KCHUNKS) {
                b[(kc + 1) & 1].p.lo = *(const int4*)(Bbuf + (kc + 1) * 32);
                b[(kc + 1) & 1].p.hi = *(const int4*)(Bbuf + (kc + 1) * 32 + 16);
            }
            acc = __builtin_amdgcn_wmma_f32_16x16x32_bf16(
                false, areg[kc].v, false, b[kc & 1].v, (short)0, acc, false, false);
        }

        #pragma unroll
        for (int i = 0; i < 8; ++i) {            // C layout: M = i + 8*(lane/16)
            const float s = acc[i] * inv;
            if (s > best[i]) { best[i] = s; bidx[i] = code; }
        }

        __syncthreads();   // everyone done reading buf before it is refilled
        if (nt + 2 < numTiles) async_copy_b_tile(embb, nt + 2, bbuf_byte[buf], tid);
    }

    // argmax reduce across the 16 lanes sharing each M row (prefer lower index on tie)
    #pragma unroll
    for (int off = 8; off >= 1; off >>= 1) {
        #pragma unroll
        for (int i = 0; i < 8; ++i) {
            const float ov = __shfl_xor(best[i], off, 32);
            const int   oi = __shfl_xor(bidx[i], off, 32);
            if (ov > best[i] || (ov == best[i] && oi < bidx[i])) { best[i] = ov; bidx[i] = oi; }
        }
    }
    if (ln == 0) {
        #pragma unroll
        for (int i = 0; i < 8; ++i) {
            const int row = m_base + i + 8 * g;
            idx_i[row]     = bidx[i];
            out_idx_f[row] = (float)bidx[i];
        }
    }
}

// ---------------- gather selected codes, commitment loss, usage counts ----------------
__global__ __launch_bounds__(256) void gather_loss_kernel(const float* __restrict__ z,
                                                          const float* __restrict__ emb,
                                                          const int* __restrict__ idx_i,
                                                          float* __restrict__ out_emb,
                                                          float* __restrict__ counts,
                                                          float* __restrict__ loss_acc) {
    const int wave = threadIdx.x >> 5;
    const int lane = threadIdx.x & 31;
    const int row  = blockIdx.x * WAVES_PER_BLOCK + wave;
    const float4* zr4 = (const float4*)(z + (size_t)row * DDIM + lane * 16);
    float4 v[4];
    float ss = 0.f;
    #pragma unroll
    for (int i = 0; i < 4; ++i) {
        v[i] = zr4[i];
        ss += v[i].x * v[i].x + v[i].y * v[i].y + v[i].z * v[i].z + v[i].w * v[i].w;
    }
    ss = wave_reduce_sum(ss);
    const float inv = 1.f / fmaxf(sqrtf(ss), 1e-12f);
    const int c = idx_i[row];
    const float4* er4 = (const float4*)(emb + (size_t)c * DDIM + lane * 16);
    float4* o4 = (float4*)(out_emb + (size_t)row * DDIM + lane * 16);
    float sq = 0.f;
    #pragma unroll
    for (int i = 0; i < 4; ++i) {
        const float4 e = er4[i];
        o4[i] = e;
        const float dx = v[i].x * inv - e.x;
        const float dy = v[i].y * inv - e.y;
        const float dz = v[i].z * inv - e.z;
        const float dw = v[i].w * inv - e.w;
        sq += dx * dx + dy * dy + dz * dz + dw * dw;
    }
    sq = wave_reduce_sum(sq);
    if (lane == 0) {
        atomicAdd(loss_acc, sq);
        atomicAdd(&counts[c], 1.0f);
    }
}

// ---------------- scalars: commitment loss + perplexity ----------------
__global__ __launch_bounds__(256) void finalize_kernel(const float* __restrict__ counts,
                                                       const float* __restrict__ loss_acc,
                                                       float* __restrict__ out_scalars,
                                                       int N, int Kc) {
    __shared__ float sh[256];
    float s = 0.f;
    for (int c = threadIdx.x; c < Kc; c += 256) {
        const float p = counts[c] / (float)N;
        s += p * logf(p + 1e-10f);
    }
    sh[threadIdx.x] = s;
    __syncthreads();
    for (int st = 128; st > 0; st >>= 1) {
        if ((int)threadIdx.x < st) sh[threadIdx.x] += sh[threadIdx.x + st];
        __syncthreads();
    }
    if (threadIdx.x == 0) {
        out_scalars[0] = 0.25f * loss_acc[0] / ((float)N * (float)DDIM);  // commitment loss
        out_scalars[1] = expf(-sh[0]);                                    // perplexity
    }
}

extern "C" void kernel_launch(void* const* d_in, const int* in_sizes, int n_in,
                              void* d_out, int out_size, void* d_ws, size_t ws_size,
                              hipStream_t stream) {
    const float* z   = (const float*)d_in[0];
    const float* emb = (const float*)d_in[1];
    const int N  = in_sizes[0] / DDIM;   // 65536
    const int Kc = in_sizes[1] / DDIM;   // 2048

    float* out         = (float*)d_out;
    float* out_emb     = out;                       // (N, D)
    float* out_idx_f   = out + (size_t)N * DDIM;    // (N,)
    float* out_scalars = out_idx_f + N;             // loss, perplexity

    char* ws = (char*)d_ws;
    unsigned short* znb      = (unsigned short*)ws;                            // N*D bf16
    unsigned short* embb     = (unsigned short*)(ws + (size_t)N * DDIM * 2);   // K*D bf16
    float*          inv_e    = (float*)(embb + (size_t)Kc * DDIM);             // K
    int*            idx_i    = (int*)(inv_e + Kc);                             // N
    float*          counts   = (float*)(idx_i + N);                            // K
    float*          loss_acc = counts + Kc;                                    // 1

    hipMemsetAsync(counts, 0, (size_t)(Kc + 1) * sizeof(float), stream);

    prep_z_kernel<<<N / WAVES_PER_BLOCK, 256, 0, stream>>>(z, znb);
    prep_e_kernel<<<Kc / WAVES_PER_BLOCK, 256, 0, stream>>>(emb, embb, inv_e);

    const size_t gemm_lds = (size_t)ROWS_PER_BLOCK * DDIM * 2 + 2 * BTILE_BYTES;  // 160 KB
    argmax_gemm_kernel<<<N / ROWS_PER_BLOCK, 256, gemm_lds, stream>>>(
        znb, embb, inv_e, out_idx_f, idx_i, Kc);

    gather_loss_kernel<<<N / WAVES_PER_BLOCK, 256, 0, stream>>>(
        z, emb, idx_i, out_emb, counts, loss_acc);
    finalize_kernel<<<1, 256, 0, stream>>>(counts, loss_acc, out_scalars, N, Kc);
}
